// One_Step_Decoder_10960756539841
// MI455X (gfx1250) — matrix operand order repaired and checked by reading, not compiled, not run on hardware
//
#include <hip/hip_runtime.h>
#include <hip/hip_bf16.h>

typedef __attribute__((ext_vector_type(16))) __bf16 v16bf;
typedef __attribute__((ext_vector_type(8)))  float  v8f;

typedef unsigned int tdm_u32x4 __attribute__((ext_vector_type(4)));
typedef int          tdm_i32x8 __attribute__((ext_vector_type(8)));
typedef int          tdm_i32x4 __attribute__((ext_vector_type(4)));

#if defined(__HIP_DEVICE_COMPILE__) && __has_builtin(__builtin_amdgcn_tensor_load_to_lds) && __has_builtin(__builtin_amdgcn_s_wait_tensorcnt)
#define HAVE_TDM 1
#else
#define HAVE_TDM 0
#endif

#define Bn   2048
#define Tn   512
#define ENCD 128
#define ATTU 64
#define VOC  2017
#define EMBD 300
#define DU   64
#define GU   192   // 3*DU

__device__ __forceinline__ float sigmoidf_(float x) {
    return 1.0f / (1.0f + __expf(-x));
}

__device__ __forceinline__ void tdm_load_tile_32x128(const float* gsrc, unsigned lds_off) {
#if HAVE_TDM
    unsigned long long ga = (unsigned long long)(size_t)gsrc;
    tdm_u32x4 g0;
    g0[0] = 1u;                                            // count=1, user mode
    g0[1] = lds_off;                                       // lds_addr
    g0[2] = (unsigned)(ga & 0xFFFFFFFFu);                  // global_addr[31:0]
    g0[3] = (unsigned)((ga >> 32) & 0x1FFFFFFu) | (2u << 30); // addr[56:32] | type=2
    tdm_i32x8 g1;
    g1[0] = (int)(2u << 16);        // data_size = 4 bytes; workgroup_mask = 0
    g1[1] = (int)(128u << 16);      // tensor_dim0 = 128 (low16)
    g1[2] = (int)(512u << 16);      // tensor_dim1 = 512 (low16)
    g1[3] = (int)(128u << 16);      // tile_dim0 = 128
    g1[4] = 32;                     // tile_dim1 = 32, tile_dim2 = 0
    g1[5] = 128;                    // tensor_dim0_stride = 128
    g1[6] = 0;
    g1[7] = 0;
    tdm_i32x4 gz = {0, 0, 0, 0};
#if __has_include(<hip/amd_detail/amd_gfx1250_TDM.h>)
    tdm_i32x8 gz8 = {0, 0, 0, 0, 0, 0, 0, 0};
    __builtin_amdgcn_tensor_load_to_lds(g0, g1, gz, gz, gz8, 0);
#else
    __builtin_amdgcn_tensor_load_to_lds(g0, g1, gz, gz, 0);
#endif
#else
    (void)gsrc; (void)lds_off;
#endif
}

// ---------------------------------------------------------------------------
// Kernel 1: fused attention. One workgroup (256 thr, 8 waves) per batch row.
//   feat  = tanh(enc[b] @ waW + wab + wbh)     WMMA bf16, 512x128 @ 128x64
//   alpha = softmax(feat @ vW + vb)            LDS
//   ctx2  = (alpha^T @ enc[b]) @ d2W + d2b     TDM-staged reduction -> ws_ctx2
// ---------------------------------------------------------------------------
__global__ __launch_bounds__(256) void attn_kernel(
    const float* __restrict__ enc,
    const float* __restrict__ for_h, const float* __restrict__ bac_h,
    const float* __restrict__ waW,  const float* __restrict__ wab,
    const float* __restrict__ wbW,  const float* __restrict__ wbb,
    const float* __restrict__ vW,   const float* __restrict__ vb,
    const float* __restrict__ d2W,  const float* __restrict__ d2b,
    float* __restrict__ out_alpha,   // (B,T)
    float* __restrict__ ws_ctx2)     // (B,300)
{
    const int b    = blockIdx.x;
    const int tid  = threadIdx.x;
    const int lane = tid & 31;
    const int wave = tid >> 5;
    const int lm   = lane & 15;
    const int half = lane >> 4;

    // frag-ready swizzled waW: [tile(kc*4+n)][lane][e], 16 KB
    __shared__ __align__(32) __bf16 sWaSw[16 * 32 * 16];
    __shared__ float  sSh[DU];
    __shared__ float  sWbh[ATTU];
    __shared__ float  sVw[ATTU];
    __shared__ float  sAlpha[Tn];
    __shared__ float  sRed[256];
    __shared__ float  sCtx[ENCD];
    __shared__ __align__(16) float sEnc[32 * ENCD];   // 16 KB TDM staging tile

    // ---- stage waW into frag-ready layout (coalesced read, LDS scatter) ----
    for (int i = tid; i < ENCD * ATTU; i += 256) {
        const int K = i >> 6, u = i & 63;
        const int n  = u >> 4, lmv = u & 15;
        const int hf = (K >> 3) & 1, e = (K & 7) + (((K >> 4) & 1) << 3), kc = K >> 5;
        sWaSw[((((kc << 2) + n) * 32) + hf * 16 + lmv) * 16 + e] = (__bf16)waW[i];
    }
    if (tid < DU) {
        sSh[tid] = for_h[b * DU + tid] + bac_h[b * DU + tid];
        sVw[tid] = vW[tid];
    }
    __syncthreads();
    if (tid < ATTU) {
        float acc = wbb[tid] + wab[tid];
        for (int k = 0; k < DU; ++k) acc += sSh[k] * wbW[k * ATTU + tid];
        sWbh[tid] = acc;
    }
    __syncthreads();
    const float vbias = vb[0];
    const v16bf* sWaFrag = (const v16bf*)sWaSw;

    // ---- phase 1: scores via WMMA; each wave owns 16-row slabs ----
    for (int t0 = wave * 16; t0 < Tn; t0 += 8 * 16) {
        const float* rowp = enc + ((size_t)b * Tn + t0 + lm) * ENCD;

        v16bf afrag[4];
        #pragma unroll
        for (int kc = 0; kc < 4; ++kc) {
            const int kb = kc * 32 + half * 8;
            #pragma unroll
            for (int e = 0; e < 8; ++e) {
                afrag[kc][e]     = (__bf16)rowp[kb + e];
                afrag[kc][8 + e] = (__bf16)rowp[kb + 16 + e];
            }
        }

        v8f acc[4];
        #pragma unroll
        for (int n = 0; n < 4; ++n) acc[n] = (v8f){};

        #pragma unroll
        for (int n = 0; n < 4; ++n) {
            #pragma unroll
            for (int kc = 0; kc < 4; ++kc) {
                const v16bf bfrag = sWaFrag[((kc << 2) + n) * 32 + lane];
                acc[n] = __builtin_amdgcn_wmma_f32_16x16x32_bf16(
                    false, afrag[kc], false, bfrag, (short)0, acc[n], false, false);
            }
        }

        float p[8];
        #pragma unroll
        for (int j = 0; j < 8; ++j) p[j] = 0.0f;
        #pragma unroll
        for (int n = 0; n < 4; ++n) {
            const int u = n * 16 + lm;
            const float bu = sWbh[u];
            const float vu = sVw[u];
            #pragma unroll
            for (int j = 0; j < 8; ++j)
                p[j] += tanhf(acc[n][j] + bu) * vu;
        }
        #pragma unroll
        for (int j = 0; j < 8; ++j) {
            float s = p[j];
            #pragma unroll
            for (int off = 1; off < 16; off <<= 1)
                s += __shfl_xor(s, off, 32);
            if (lm == 0) sAlpha[t0 + j + half * 8] = s + vbias;
        }
    }
    __syncthreads();

    // ---- phase 2: softmax over T in LDS ----
    float m = -3.402823466e38f;
    for (int t = tid; t < Tn; t += 256) m = fmaxf(m, sAlpha[t]);
    sRed[tid] = m; __syncthreads();
    for (int s = 128; s > 0; s >>= 1) {
        if (tid < s) sRed[tid] = fmaxf(sRed[tid], sRed[tid + s]);
        __syncthreads();
    }
    m = sRed[0]; __syncthreads();

    float sum = 0.0f;
    for (int t = tid; t < Tn; t += 256) {
        float e = __expf(sAlpha[t] - m);
        sAlpha[t] = e;
        sum += e;
    }
    sRed[tid] = sum; __syncthreads();
    for (int s = 128; s > 0; s >>= 1) {
        if (tid < s) sRed[tid] += sRed[tid + s];
        __syncthreads();
    }
    const float inv = 1.0f / sRed[0];
    __syncthreads();
    for (int t = tid; t < Tn; t += 256) {
        float a = sAlpha[t] * inv;
        sAlpha[t] = a;
        out_alpha[(size_t)b * Tn + t] = a;
    }
    __syncthreads();

    // ---- phase 3: context = sum_t alpha_t * enc[b,t,:] ----
#if HAVE_TDM
    {
        const unsigned lds_off = (unsigned)(size_t)(void*)sEnc;  // low 32 bits = LDS offset
        float cacc = 0.0f;
        for (int c = 0; c < Tn / 32; ++c) {
            if (tid == 0) {
                tdm_load_tile_32x128(enc + ((size_t)b * Tn + c * 32) * ENCD, lds_off);
                __builtin_amdgcn_s_wait_tensorcnt(0);
            }
            __syncthreads();
            if (tid < ENCD) {
                #pragma unroll 4
                for (int r = 0; r < 32; ++r)
                    cacc += sAlpha[c * 32 + r] * sEnc[r * ENCD + tid];
            }
            __syncthreads();
        }
        if (tid < ENCD) sCtx[tid] = cacc;
    }
#else
    if (tid < ENCD) {
        float cacc = 0.0f;
        const float* ep = enc + (size_t)b * Tn * ENCD + tid;
        for (int t = 0; t < Tn; ++t) cacc += sAlpha[t] * ep[(size_t)t * ENCD];
        sCtx[tid] = cacc;
    }
#endif
    __syncthreads();

    // ---- phase 4: ctx2 = context @ d2W + d2b ----
    for (int u = tid; u < EMBD; u += 256) {
        float a = d2b[u];
        for (int k = 0; k < ENCD; ++k) a += sCtx[k] * d2W[k * EMBD + u];
        ws_ctx2[(size_t)b * EMBD + u] = a;
    }
}

// ---------------------------------------------------------------------------
// Kernel 2: embedding gather + both GRU steps. One workgroup per batch row.
// ---------------------------------------------------------------------------
__global__ __launch_bounds__(GU) void gru_kernel(
    const int*   __restrict__ tokens,
    const float* __restrict__ emb_table,
    const float* __restrict__ ws_ctx2,
    const float* __restrict__ for_h, const float* __restrict__ bac_h,
    const float* __restrict__ fK, const float* __restrict__ fR, const float* __restrict__ fb,
    const float* __restrict__ bK, const float* __restrict__ bR, const float* __restrict__ bb,
    float* __restrict__ out_fwd, float* __restrict__ out_bwd,
    float* __restrict__ ws_hcat)     // (B,128) = [fwd|bwd]
{
    const int b   = blockIdx.x;
    const int tid = threadIdx.x;     // 0..191

    __shared__ float sX[2 * EMBD];
    __shared__ float sHf[DU], sHb[DU];
    __shared__ float sXZf[GU], sHZf[GU], sXZb[GU], sHZb[GU];

    const int tok = tokens[b];
    for (int i = tid; i < EMBD; i += GU) {
        sX[i]        = ws_ctx2[(size_t)b * EMBD + i];
        sX[EMBD + i] = emb_table[(size_t)tok * EMBD + i];
    }
    if (tid < DU) {
        sHf[tid] = for_h[b * DU + tid];
        sHb[tid] = bac_h[b * DU + tid];
    }
    __syncthreads();

    {
        float axf = fb[tid];
        float axb = bb[tid];
        for (int k = 0; k < 2 * EMBD; ++k) {
            const float x = sX[k];
            axf += x * fK[k * GU + tid];
            axb += x * bK[k * GU + tid];
        }
        float ahf = fb[GU + tid];
        float ahb = bb[GU + tid];
        for (int k = 0; k < DU; ++k) {
            ahf += sHf[k] * fR[k * GU + tid];
            ahb += sHb[k] * bR[k * GU + tid];
        }
        sXZf[tid] = axf; sHZf[tid] = ahf;
        sXZb[tid] = axb; sHZb[tid] = ahb;
    }
    __syncthreads();

    if (tid < DU) {
        float z  = sigmoidf_(sXZf[tid]        + sHZf[tid]);
        float r  = sigmoidf_(sXZf[DU + tid]   + sHZf[DU + tid]);
        float hh = tanhf    (sXZf[2*DU + tid] + r * sHZf[2*DU + tid]);
        float h  = z * sHf[tid] + (1.0f - z) * hh;
        out_fwd[b * DU + tid]          = h;
        ws_hcat[(size_t)b * 128 + tid] = h;

        z  = sigmoidf_(sXZb[tid]        + sHZb[tid]);
        r  = sigmoidf_(sXZb[DU + tid]   + sHZb[DU + tid]);
        hh = tanhf    (sXZb[2*DU + tid] + r * sHZb[2*DU + tid]);
        h  = z * sHb[tid] + (1.0f - z) * hh;
        out_bwd[b * DU + tid]               = h;
        ws_hcat[(size_t)b * 128 + DU + tid] = h;
    }
}

// ---------------------------------------------------------------------------
// Kernel 3: dense_op = hcat(2048x128) @ out_W(128x2017) + out_b  (WMMA bf16)
//   grid (16, 16): each block = 128 rows x 128 cols; both operand panels
//   staged once in frag-ready LDS; 8 waves x 8 M-tiles x 4 K-steps of WMMA.
// ---------------------------------------------------------------------------
__global__ __launch_bounds__(256) void dense_kernel(
    const float* __restrict__ ws_hcat,
    const float* __restrict__ outW, const float* __restrict__ outb,
    float* __restrict__ out)
{
    const int tid  = threadIdx.x;
    const int lane = tid & 31;
    const int wave = tid >> 5;
    const int lm   = lane & 15;
    const int half = lane >> 4;
    const int m0   = blockIdx.x * 128;
    const int n0   = blockIdx.y * 128;

    __shared__ __align__(32) __bf16 sWb[8 * 4 * 32 * 16];  // 32 KB: [wave][kc][lane][e]
    __shared__ __align__(32) __bf16 sHa[8 * 4 * 32 * 16];  // 32 KB: [mt][kc][lane][e]

    // stage out_W panel (zero-pad n >= VOC so WMMA runs with EXEC all-1s)
    for (int i = tid; i < 128 * 128; i += 256) {
        const int K = i >> 7, nc = i & 127;
        const int n = n0 + nc;
        const float v = (n < VOC) ? outW[(size_t)K * VOC + n] : 0.0f;
        const int wv = nc >> 4, lmv = nc & 15;
        const int hf = (K >> 3) & 1, e = (K & 7) + (((K >> 4) & 1) << 3), kc = K >> 5;
        sWb[((((wv << 2) + kc) * 32) + hf * 16 + lmv) * 16 + e] = (__bf16)v;
    }
    // stage H rows in A-fragment layout
    for (int i = tid; i < 128 * 128; i += 256) {
        const int mr = i >> 7, K = i & 127;
        const float v = ws_hcat[(size_t)(m0 + mr) * 128 + K];
        const int mt = mr >> 4, mm = mr & 15;
        const int kc = K >> 5, kk = K & 31;
        const int hf = (kk >> 3) & 1;
        const int e  = (kk & 7) + ((kk >> 4) << 3);
        sHa[((((mt << 2) + kc) * 32) + hf * 16 + mm) * 16 + e] = (__bf16)v;
    }
    __syncthreads();

    const v16bf* fWb = (const v16bf*)sWb;
    const v16bf* fHa = (const v16bf*)sHa;

    v16bf bfrag[4];
    #pragma unroll
    for (int kc = 0; kc < 4; ++kc)
        bfrag[kc] = fWb[((wave << 2) + kc) * 32 + lane];

    const int n = n0 + wave * 16 + lm;
    const bool nvalid = (n < VOC);
    const float bias = nvalid ? outb[n] : 0.0f;

    for (int mt = 0; mt < 8; ++mt) {
        v16bf a[4];
        #pragma unroll
        for (int kc = 0; kc < 4; ++kc)
            a[kc] = fHa[((mt << 2) + kc) * 32 + lane];

        v8f acc = (v8f){};
        #pragma unroll
        for (int kc = 0; kc < 4; ++kc)
            acc = __builtin_amdgcn_wmma_f32_16x16x32_bf16(
                false, a[kc], false, bfrag[kc], (short)0, acc, false, false);

        if (nvalid) {
            #pragma unroll
            for (int j = 0; j < 8; ++j) {
                const int mrow = m0 + mt * 16 + j + half * 8;
                out[(size_t)mrow * VOC + n] = acc[j] + bias;
            }
        }
    }
}

// ---------------------------------------------------------------------------
extern "C" void kernel_launch(void* const* d_in, const int* in_sizes, int n_in,
                              void* d_out, int out_size, void* d_ws, size_t ws_size,
                              hipStream_t stream) {
    const int*   tokens = (const int*)  d_in[0];
    const float* enc    = (const float*)d_in[1];
    const float* for_h  = (const float*)d_in[2];
    const float* bac_h  = (const float*)d_in[3];
    const float* emb    = (const float*)d_in[4];
    const float* waW    = (const float*)d_in[5];
    const float* wab    = (const float*)d_in[6];
    const float* wbW    = (const float*)d_in[7];
    const float* wbb    = (const float*)d_in[8];
    const float* vW     = (const float*)d_in[9];
    const float* vb     = (const float*)d_in[10];
    const float* d2W    = (const float*)d_in[11];
    const float* d2b    = (const float*)d_in[12];
    const float* fK     = (const float*)d_in[13];
    const float* fR     = (const float*)d_in[14];
    const float* fb     = (const float*)d_in[15];
    const float* bK     = (const float*)d_in[16];
    const float* bR     = (const float*)d_in[17];
    const float* bb     = (const float*)d_in[18];
    const float* outW   = (const float*)d_in[19];
    const float* outb   = (const float*)d_in[20];

    float* out       = (float*)d_out;
    float* out_dense = out;                                  // B*VOC
    float* out_fwd   = out_dense + (size_t)Bn * VOC;         // B*DU
    float* out_bwd   = out_fwd   + (size_t)Bn * DU;          // B*DU
    float* out_alpha = out_bwd   + (size_t)Bn * DU;          // B*T

    float* ws_ctx2 = (float*)d_ws;                           // B*300
    float* ws_hcat = ws_ctx2 + (size_t)Bn * EMBD;            // B*128

    attn_kernel<<<Bn, 256, 0, stream>>>(enc, for_h, bac_h, waW, wab, wbW, wbb,
                                        vW, vb, d2W, d2b, out_alpha, ws_ctx2);
    gru_kernel<<<Bn, GU, 0, stream>>>(tokens, emb, ws_ctx2, for_h, bac_h,
                                      fK, fR, fb, bK, bR, bb,
                                      out_fwd, out_bwd, ws_hcat);
    dense_kernel<<<dim3(16, 16), 256, 0, stream>>>(ws_hcat, outW, outb, out_dense);
}
